// MathFormer_8632884265125
// MI455X (gfx1250) — matrix operand order repaired and checked by reading, not compiled, not run on hardware
//
#include <hip/hip_runtime.h>
#include <hip/hip_bf16.h>
#include <math.h>

// ---------------------------------------------------------------------------
// GPT-2 style forward for MI455X (gfx1250, wave32, WMMA).
// GEMMs and attention both run on v_wmma_f32_16x16x32_bf16 (bf16 operands,
// fp32 accumulation). LN / softmax statistics in fp32.
// ---------------------------------------------------------------------------

#define BB 2
#define TT 1024
#define DD 768
#define HH 12
#define HD 64
#define LL 6
#define VV 32000
#define MM (BB * TT)   // 2048 rows of activations
#define QB 8           // query blocks per sequence (TT / 128)

typedef __attribute__((ext_vector_type(16))) __bf16 v16bf;
typedef __attribute__((ext_vector_type(8)))  float  v8f;

union Frag { uint4 q[2]; unsigned int d[8]; v16bf v; };

__device__ __forceinline__ unsigned int pack_bf16(float a, float b) {
    union { __bf16 h[2]; unsigned int u; } p;
    p.h[0] = (__bf16)a;
    p.h[1] = (__bf16)b;
    return p.u;   // lowers to v_cvt_pk_bf16_f32
}

// ---------------------------------------------------------------------------
// Embedding: x[b,t,:] = wte[idx[b,t],:] + wpe[t,:]
// ---------------------------------------------------------------------------
__global__ __launch_bounds__(256) void gpt_embed_kernel(
    const int* __restrict__ idx, const float* __restrict__ wte,
    const float* __restrict__ wpe, float* __restrict__ x, int total)
{
    int i = blockIdx.x * 256 + threadIdx.x;
    if (i >= total) return;
    int d  = i % DD;
    int bt = i / DD;
    int t  = bt % TT;
    x[i] = wte[(size_t)idx[bt] * DD + d] + wpe[(size_t)t * DD + d];
}

// ---------------------------------------------------------------------------
// LayerNorm over last dim (DD), one row per block.
// ---------------------------------------------------------------------------
__global__ __launch_bounds__(256) void gpt_ln_kernel(
    const float* __restrict__ x, const float* __restrict__ w,
    const float* __restrict__ b, float* __restrict__ out)
{
    __shared__ float red[256];
    const int row = blockIdx.x;
    const float* xr = x + (size_t)row * DD;

    float s = 0.f;
    for (int i = threadIdx.x; i < DD; i += 256) s += xr[i];
    red[threadIdx.x] = s;
    __syncthreads();
    for (int o = 128; o; o >>= 1) {
        if (threadIdx.x < o) red[threadIdx.x] += red[threadIdx.x + o];
        __syncthreads();
    }
    const float mu = red[0] * (1.0f / DD);
    __syncthreads();

    float vs = 0.f;
    for (int i = threadIdx.x; i < DD; i += 256) {
        float d = xr[i] - mu;
        vs += d * d;
    }
    red[threadIdx.x] = vs;
    __syncthreads();
    for (int o = 128; o; o >>= 1) {
        if (threadIdx.x < o) red[threadIdx.x] += red[threadIdx.x + o];
        __syncthreads();
    }
    const float rstd = rsqrtf(red[0] * (1.0f / DD) + 1e-5f);

    float* orow = out + (size_t)row * DD;
    for (int i = threadIdx.x; i < DD; i += 256)
        orow[i] = (xr[i] - mu) * rstd * w[i] + b[i];
}

// ---------------------------------------------------------------------------
// WMMA GEMM: out[M,N] = epilogue(A[M,K] @ W[K,N] + bias)
//   EPI 0: + bias (bias may be null)
//   EPI 1: gelu(. + bias)            (exact gelu, erf)
//   EPI 2: (. + bias) + resid[M,N]
// M % 128 == 0, N % 128 == 0, K % 32 == 0 (all satisfied here).
// Block: 256 threads = 8 waves; block tile 128x128; wave tile 32x64.
// LDS holds bf16 pairs, padded stride 20 dwords -> 16B-aligned fragment
// quads (ds_load_b128) and near-conflict-free banking.
// ---------------------------------------------------------------------------
template <int EPI>
__global__ __launch_bounds__(256) void gpt_gemm_wmma_kernel(
    const float* __restrict__ A, const float* __restrict__ W,
    const float* __restrict__ bias, const float* __restrict__ resid,
    float* __restrict__ out, int M, int N, int K)
{
    __shared__ unsigned int AsP[128][20];  // [row][k-pair]
    __shared__ unsigned int BsP[128][20];  // [col][k-pair]

    const int tid  = threadIdx.x;
    const int lane = tid & 31;
    const int wave = tid >> 5;
    const int bm   = blockIdx.y * 128;
    const int bn   = blockIdx.x * 128;
    const int wm   = (wave & 3) * 32;   // 4 waves along M
    const int wn   = (wave >> 2) * 64;  // 2 waves along N
    const int g    = lane >> 4;         // lane group (ISA 7.12.2)
    const int mr   = lane & 15;

    v8f acc[2][4] = {};

    for (int k0 = 0; k0 < K; k0 += 32) {
        if (k0 + 32 < K) {  // prefetch next K-tiles -> global_prefetch_b8
            __builtin_prefetch(W + (size_t)(k0 + 32) * N + bn + (tid >> 1), 0, 0);
            __builtin_prefetch(A + (size_t)(bm + (tid >> 1)) * K + (k0 + 32), 0, 0);
        }

        // Stage A: 128 rows x 16 k-pairs, 8 dwords/thread.
        #pragma unroll
        for (int it = 0; it < 8; ++it) {
            const int pi  = tid + it * 256;
            const int row = pi >> 4;
            const int p   = pi & 15;
            const float2 a2 =
                *(const float2*)(A + (size_t)(bm + row) * K + k0 + 2 * p);
            AsP[row][p] = pack_bf16(a2.x, a2.y);
        }
        // Stage B (transposed per lane): 128 cols x 16 k-pairs.
        #pragma unroll
        for (int it = 0; it < 8; ++it) {
            const int pi  = tid + it * 256;
            const int col = pi & 127;
            const int kp  = pi >> 7;
            const float b0 = W[(size_t)(k0 + 2 * kp)     * N + bn + col];
            const float b1 = W[(size_t)(k0 + 2 * kp + 1) * N + bn + col];
            BsP[col][kp] = pack_bf16(b0, b1);
        }
        __syncthreads();

        // A: element e -> k = e+8g (e<8), e+8+8g (e>=8)
        v16bf afrag[2];
        #pragma unroll
        for (int mt = 0; mt < 2; ++mt) {
            const int row = wm + mt * 16 + mr;
            Frag f;
            f.q[0] = *(const uint4*)&AsP[row][4 * g];
            f.q[1] = *(const uint4*)&AsP[row][8 + 4 * g];
            afrag[mt] = f.v;
        }
        // B (column-major per lane): element e -> k = e+8g (e<8), e+8+8g (e>=8)
        v16bf bfrag[4];
        #pragma unroll
        for (int nt = 0; nt < 4; ++nt) {
            const int col = wn + nt * 16 + mr;
            Frag f;
            f.q[0] = *(const uint4*)&BsP[col][4 * g];
            f.q[1] = *(const uint4*)&BsP[col][8 + 4 * g];
            bfrag[nt] = f.v;
        }

        #pragma unroll
        for (int mt = 0; mt < 2; ++mt)
            #pragma unroll
            for (int nt = 0; nt < 4; ++nt)
                acc[mt][nt] = __builtin_amdgcn_wmma_f32_16x16x32_bf16(
                    false, afrag[mt], false, bfrag[nt],
                    (short)0, acc[mt][nt], false, false);
        __syncthreads();
    }

    // Epilogue. C/D layout: VGPR r, lane group g -> M = r + 8*g; N = lane&15.
    #pragma unroll
    for (int mt = 0; mt < 2; ++mt) {
        #pragma unroll
        for (int nt = 0; nt < 4; ++nt) {
            const int gn = bn + wn + nt * 16 + mr;
            const float bv = bias ? bias[gn] : 0.f;
            #pragma unroll
            for (int r = 0; r < 8; ++r) {
                const int gm = bm + wm + mt * 16 + r + 8 * g;
                float v = acc[mt][nt][r] + bv;
                if (EPI == 1) v = 0.5f * v * (1.0f + erff(v * 0.70710678118654752f));
                if (EPI == 2) v += resid[(size_t)gm * N + gn];
                out[(size_t)gm * N + gn] = v;
            }
        }
    }
}

// ---------------------------------------------------------------------------
// WMMA flash attention. One block = (b, h, 128-query strip); 8 waves x 16
// queries. Computes S^T = K @ Q^T so softmax stats live on the lane axis
// (query) and the S^T register layout IS the A-fragment layout for the
// following P @ V WMMA (two 16-key tiles pack into one 16x32 P operand).
// qkv rows are [q | k | v] (3*DD); y written in [b,t, h*HD+d] layout.
// ---------------------------------------------------------------------------
__global__ __launch_bounds__(256) void gpt_attn_wmma_kernel(
    const float* __restrict__ qkv, float* __restrict__ y)
{
    __shared__ unsigned int QsP[128][36];  // [query][dim-pair 0..31]
    __shared__ unsigned int KsP[32][36];   // [key]  [dim-pair 0..31]
    __shared__ unsigned int VsP[64][20];   // [dim]  [key-pair 0..15]

    const int bid  = blockIdx.x;
    const int qblk = bid & (QB - 1);
    const int h    = (bid / QB) % HH;
    const int b    = bid / (QB * HH);
    const int q0   = qblk * 128;            // first query of this block

    const int tid  = threadIdx.x;
    const int lane = tid & 31;
    const int wave = tid >> 5;
    const int g    = lane >> 4;
    const int mr   = lane & 15;
    const int qw   = q0 + wave * 16;        // wave's first query
    const float scale = 0.125f;             // 1/sqrt(HD)

    // Stage Q strip once: 128 queries x 32 dim-pairs.
    #pragma unroll
    for (int it = 0; it < 16; ++it) {
        const int pi = tid + it * 256;
        const int qq = pi >> 5;
        const int p  = pi & 31;
        const float2 a2 = *(const float2*)(
            qkv + (size_t)(b * TT + q0 + qq) * (3 * DD) + h * HD + 2 * p);
        QsP[qq][p] = pack_bf16(a2.x, a2.y);
    }
    __syncthreads();

    // Q as B-fragments (lane = query, element e -> dim = e+8g (e<8), e+8+8g (e>=8), + 32c).
    v16bf qfrag[2];
    #pragma unroll
    for (int c = 0; c < 2; ++c) {
        Frag f;
        f.q[0] = *(const uint4*)&QsP[wave * 16 + mr][4 * g + 16 * c];
        f.q[1] = *(const uint4*)&QsP[wave * 16 + mr][8 + 4 * g + 16 * c];
        qfrag[c] = f.v;
    }

    v8f oacc[4] = {};
    float mx = -1e30f, lsum = 0.f;
    const int q_glob = qw + mr;             // this lane's query (softmax axis)

    const int kend = q0 + 128;              // keys needed by this block
    for (int kb = 0; kb < kend; kb += 32) {
        // Stage K tile: 32 keys x 32 dim-pairs.
        #pragma unroll
        for (int it = 0; it < 4; ++it) {
            const int pi = tid + it * 256;
            const int kk = pi >> 5;
            const int p  = pi & 31;
            const float2 a2 = *(const float2*)(
                qkv + (size_t)(b * TT + kb + kk) * (3 * DD) + DD + h * HD + 2 * p);
            KsP[kk][p] = pack_bf16(a2.x, a2.y);
        }
        // Stage V tile transposed: 64 dims x 16 key-pairs.
        #pragma unroll
        for (int it = 0; it < 4; ++it) {
            const int pi = tid + it * 256;
            const int dd = pi & 63;
            const int kp = pi >> 6;
            const float v0 = qkv[(size_t)(b * TT + kb + 2 * kp)     * (3 * DD) + 2 * DD + h * HD + dd];
            const float v1 = qkv[(size_t)(b * TT + kb + 2 * kp + 1) * (3 * DD) + 2 * DD + h * HD + dd];
            VsP[dd][kp] = pack_bf16(v0, v1);
        }
        __syncthreads();

        if (kb <= qw + 15) {  // wave-uniform: tile has unmasked keys
            // S^T = K @ Q^T : rows = keys, cols = queries (lane axis).
            v8f s0 = {}, s1 = {};
            #pragma unroll
            for (int c = 0; c < 2; ++c) {
                Frag f;
                f.q[0] = *(const uint4*)&KsP[mr][4 * g + 16 * c];
                f.q[1] = *(const uint4*)&KsP[mr][8 + 4 * g + 16 * c];
                s0 = __builtin_amdgcn_wmma_f32_16x16x32_bf16(
                    false, f.v, false, qfrag[c], (short)0, s0, false, false);
                f.q[0] = *(const uint4*)&KsP[16 + mr][4 * g + 16 * c];
                f.q[1] = *(const uint4*)&KsP[16 + mr][8 + 4 * g + 16 * c];
                s1 = __builtin_amdgcn_wmma_f32_16x16x32_bf16(
                    false, f.v, false, qfrag[c], (short)0, s1, false, false);
            }
            // Scale + causal mask (reg r, group g -> key kb + r + 8g [+16]).
            float p0[8], p1[8];
            float tmax = -1e30f;
            #pragma unroll
            for (int r = 0; r < 8; ++r) {
                const int k0g = kb + r + 8 * g;
                const int k1g = k0g + 16;
                p0[r] = (k0g <= q_glob) ? s0[r] * scale : -1e30f;
                p1[r] = (k1g <= q_glob) ? s1[r] * scale : -1e30f;
                tmax = fmaxf(tmax, fmaxf(p0[r], p1[r]));
            }
            tmax = fmaxf(tmax, __shfl_xor(tmax, 16, 32));
            const float mnew = fmaxf(mx, tmax);
            const float corr = __expf(mx - mnew);
            float ls = 0.f;
            #pragma unroll
            for (int r = 0; r < 8; ++r) {
                p0[r] = __expf(p0[r] - mnew);
                p1[r] = __expf(p1[r] - mnew);
                ls += p0[r] + p1[r];
            }
            ls += __shfl_xor(ls, 16, 32);
            lsum = lsum * corr + ls;
            mx = mnew;

            // P is already in A-fragment order: e<8 -> tile0 reg e,
            // e>=8 -> tile1 reg e-8 (keys {0-7,16-23}|g=0, {8-15,24-31}|g=1).
            Frag pf;
            #pragma unroll
            for (int d = 0; d < 4; ++d) pf.d[d]     = pack_bf16(p0[2 * d], p0[2 * d + 1]);
            #pragma unroll
            for (int d = 0; d < 4; ++d) pf.d[4 + d] = pack_bf16(p1[2 * d], p1[2 * d + 1]);

            // Broadcast per-query corr to O layout (query on reg axis).
            float crr[8];
            #pragma unroll
            for (int r = 0; r < 8; ++r) crr[r] = __shfl(corr, r + 8 * g, 32);
            #pragma unroll
            for (int nt = 0; nt < 4; ++nt) {
                #pragma unroll
                for (int r = 0; r < 8; ++r) oacc[nt][r] *= crr[r];
                Frag vf;
                vf.q[0] = *(const uint4*)&VsP[nt * 16 + mr][4 * g];
                vf.q[1] = *(const uint4*)&VsP[nt * 16 + mr][8 + 4 * g];
                oacc[nt] = __builtin_amdgcn_wmma_f32_16x16x32_bf16(
                    false, pf.v, false, vf.v, (short)0, oacc[nt], false, false);
            }
        }
        __syncthreads();
    }

    // Final 1/l scaling + store. O layout: lane = dim col, reg r -> query.
    const float linv = 1.0f / lsum;
    float lr[8];
    #pragma unroll
    for (int r = 0; r < 8; ++r) lr[r] = __shfl(linv, r + 8 * g, 32);
    #pragma unroll
    for (int nt = 0; nt < 4; ++nt) {
        #pragma unroll
        for (int r = 0; r < 8; ++r) {
            const int q = qw + r + 8 * g;
            y[(size_t)(b * TT + q) * DD + h * HD + nt * 16 + mr] = oacc[nt][r] * lr[r];
        }
    }
}

// ---------------------------------------------------------------------------
// Host-side orchestration.
// Inputs (setup_inputs order):
//  0 idx  1 wte  2 wpe  3 ln1_w 4 ln1_b 5 attn_w 6 attn_b 7 proj_w 8 proj_b
//  9 ln2_w 10 ln2_b 11 fc_w 12 fc_b 13 fcproj_w 14 fcproj_b
//  15 lnf_w 16 lnf_b 17 head_w
// ---------------------------------------------------------------------------
extern "C" void kernel_launch(void* const* d_in, const int* in_sizes, int n_in,
                              void* d_out, int out_size, void* d_ws, size_t ws_size,
                              hipStream_t stream)
{
    (void)in_sizes; (void)n_in; (void)out_size; (void)ws_size;

    const int*   idx      = (const int*)  d_in[0];
    const float* wte      = (const float*)d_in[1];
    const float* wpe      = (const float*)d_in[2];
    const float* ln1_w    = (const float*)d_in[3];
    const float* ln1_b    = (const float*)d_in[4];
    const float* attn_w   = (const float*)d_in[5];
    const float* attn_b   = (const float*)d_in[6];
    const float* proj_w   = (const float*)d_in[7];
    const float* proj_b   = (const float*)d_in[8];
    const float* ln2_w    = (const float*)d_in[9];
    const float* ln2_b    = (const float*)d_in[10];
    const float* fc_w     = (const float*)d_in[11];
    const float* fc_b     = (const float*)d_in[12];
    const float* fcproj_w = (const float*)d_in[13];
    const float* fcproj_b = (const float*)d_in[14];
    const float* lnf_w    = (const float*)d_in[15];
    const float* lnf_b    = (const float*)d_in[16];
    const float* head_w   = (const float*)d_in[17];
    float*       logits   = (float*)d_out;

    // Workspace carve-up (floats).
    float* ws   = (float*)d_ws;
    float* x    = ws;                          // [MM, DD]
    float* hbuf = x    + (size_t)MM * DD;      // [MM, DD]   (LN out, reused)
    float* qkv  = hbuf + (size_t)MM * DD;      // [MM, 3*DD]
    float* ybuf = qkv  + (size_t)MM * 3 * DD;  // [MM, DD]
    float* mbuf = ybuf + (size_t)MM * DD;      // [MM, 4*DD]

    const dim3 blk(256);

    // Embedding
    {
        int total = MM * DD;
        gpt_embed_kernel<<<(total + 255) / 256, blk, 0, stream>>>(idx, wte, wpe, x, total);
    }

    // Transformer blocks
    for (int l = 0; l < LL; ++l) {
        const float* l1w = ln1_w    + (size_t)l * DD;
        const float* l1b = ln1_b    + (size_t)l * DD;
        const float* aw  = attn_w   + (size_t)l * DD * 3 * DD;
        const float* ab  = attn_b   + (size_t)l * 3 * DD;
        const float* pw  = proj_w   + (size_t)l * DD * DD;
        const float* pb  = proj_b   + (size_t)l * DD;
        const float* l2w = ln2_w    + (size_t)l * DD;
        const float* l2b = ln2_b    + (size_t)l * DD;
        const float* fw  = fc_w     + (size_t)l * DD * 4 * DD;
        const float* fb  = fc_b     + (size_t)l * 4 * DD;
        const float* fpw = fcproj_w + (size_t)l * 4 * DD * DD;
        const float* fpb = fcproj_b + (size_t)l * DD;

        // h = LN1(x)
        gpt_ln_kernel<<<MM, blk, 0, stream>>>(x, l1w, l1b, hbuf);
        // qkv = h @ aw + ab            [2048 x 2304], K=768
        gpt_gemm_wmma_kernel<0><<<dim3(3 * DD / 128, MM / 128), blk, 0, stream>>>(
            hbuf, aw, ab, nullptr, qkv, MM, 3 * DD, DD);
        // y = softmax(q k^T / sqrt(hd), causal) v   (WMMA flash attention)
        gpt_attn_wmma_kernel<<<BB * HH * QB, blk, 0, stream>>>(qkv, ybuf);
        // x = x + y @ pw + pb          [2048 x 768], K=768
        gpt_gemm_wmma_kernel<2><<<dim3(DD / 128, MM / 128), blk, 0, stream>>>(
            ybuf, pw, pb, x, x, MM, DD, DD);
        // h = LN2(x)
        gpt_ln_kernel<<<MM, blk, 0, stream>>>(x, l2w, l2b, hbuf);
        // m = gelu(h @ fw + fb)        [2048 x 3072], K=768
        gpt_gemm_wmma_kernel<1><<<dim3(4 * DD / 128, MM / 128), blk, 0, stream>>>(
            hbuf, fw, fb, nullptr, mbuf, MM, 4 * DD, DD);
        // x = x + m @ fpw + fpb        [2048 x 768], K=3072
        gpt_gemm_wmma_kernel<2><<<dim3(DD / 128, MM / 128), blk, 0, stream>>>(
            mbuf, fpw, fpb, x, x, MM, DD, 4 * DD);
    }

    // Final LN + head: logits = LN_f(x) @ head_w   [2048 x 32000], K=768
    gpt_ln_kernel<<<MM, blk, 0, stream>>>(x, lnf_w, lnf_b, hbuf);
    gpt_gemm_wmma_kernel<0><<<dim3(VV / 128, MM / 128), blk, 0, stream>>>(
        hbuf, head_w, nullptr, nullptr, logits, MM, VV, DD);
}
